// HybridBandSelectionModel_5420248727584
// MI455X (gfx1250) — matrix operand order repaired and checked by reading, not compiled, not run on hardware
//
#include <hip/hip_runtime.h>
#include <math.h>

typedef _Float16 v16h __attribute__((ext_vector_type(16)));
typedef _Float16 v8h  __attribute__((ext_vector_type(8)));
typedef float    v8f  __attribute__((ext_vector_type(8)));

#define C_BANDS 400
#define LDH     416   // padded band dim (13 * 32)
#define KSEL    10
#define LATENT  8
#define FFDIM   128

// ---------------- weight pack: f32 [P,Q] -> f16 [PP,KP] zero-padded ----------------
__global__ void k_wpack(const float* __restrict__ W, _Float16* __restrict__ Wh,
                        int P, int Q, int KP) {
  const int idx = blockIdx.x * 256 + threadIdx.x;
  const int r = idx / KP;
  const int k = idx - r * KP;
  _Float16 v = (_Float16)0.f;
  if (r < P && k < Q) v = (_Float16)W[(size_t)r * Q + k];
  Wh[idx] = v;
}

// ---------------- transpose [C,N] -> [N,LDH] with positional add, zero pad ----------
__global__ void k_transpose_pos(const float* __restrict__ x, const float* __restrict__ pos,
                                float* __restrict__ h, int Ntok) {
  __shared__ float tile[32][33];
  const int pBase = blockIdx.x * 32;
  const int cBase = blockIdx.y * 32;   // grid.y = LDH/32 = 13 -> covers 0..415
  const int tx = threadIdx.x, ty = threadIdx.y;
#pragma unroll
  for (int j = 0; j < 4; ++j) {
    int c = cBase + ty + j * 8;
    float v = 0.f;
    if (c < C_BANDS) v = x[(size_t)c * Ntok + pBase + tx];
    tile[ty + j * 8][tx] = v;
  }
  __syncthreads();
#pragma unroll
  for (int j = 0; j < 4; ++j) {
    int c = cBase + tx;
    int p = pBase + ty + j * 8;
    float v = (c < C_BANDS) ? (tile[tx][ty + j * 8] + pos[c]) : 0.f;
    h[(size_t)p * LDH + c] = v;
  }
}

// ---------------- WMMA f16 GEMM: Y[N,PP] = act(X[N,KP] @ Wh[PP,KP]^T + bias (+res)) --
// Block: 256 threads = 8 waves (4 row x 2 col); wave tile 32x32; block tile 128x64.
// K stepped by 32 with double-buffered LDS; activations f32->f16 on staging.
__global__ void k_gemm_wmma(const float* __restrict__ X, const _Float16* __restrict__ Wh,
                            const float* __restrict__ bias, const float* __restrict__ res,
                            float* __restrict__ Y, int P, int PP, int KP, int relu) {
  __shared__ __align__(16) _Float16 As[2][128][40];  // 80B rows: 16B-aligned b128 reads
  __shared__ __align__(16) _Float16 Bs[2][64][40];

  const int tid  = threadIdx.x;
  const int lane = tid & 31;
  const int wave = tid >> 5;
  const int wr   = wave >> 1;           // 0..3 : 32-row subtile
  const int wc   = wave & 1;            // 0..1 : 32-col subtile
  const int rowBase = blockIdx.x * 128;
  const int colBase = blockIdx.y * 64;
  const int m  = lane & 15;
  const int hi = lane >> 4;
  const int KT = KP >> 5;

  auto stage = [&](int kt, int buf) {
    const int k0 = kt << 5;
    {  // A: 128 rows x 32 k (f32 -> f16), 16 halfs per thread, vectorized
      const int r  = tid >> 1;
      const int c0 = (tid & 1) << 4;
      const float4* s4 = (const float4*)(X + (size_t)(rowBase + r) * KP + k0 + c0);
      _Float16 tmp[16];
#pragma unroll
      for (int j = 0; j < 4; ++j) {
        float4 f = s4[j];
        tmp[j * 4 + 0] = (_Float16)f.x; tmp[j * 4 + 1] = (_Float16)f.y;
        tmp[j * 4 + 2] = (_Float16)f.z; tmp[j * 4 + 3] = (_Float16)f.w;
      }
      *(v8h*)&As[buf][r][c0]     = *(v8h*)&tmp[0];
      *(v8h*)&As[buf][r][c0 + 8] = *(v8h*)&tmp[8];
    }
    {  // B: 64 cols x 32 k from padded f16 weights, one b128 per thread
      const int c  = tid >> 2;
      const int c0 = (tid & 3) << 3;
      const int gc = colBase + c;
      v8h v = {};
      if (gc < PP) v = *(const v8h*)&Wh[(size_t)gc * KP + k0 + c0];
      *(v8h*)&Bs[buf][c][c0] = v;
    }
  };

  v8f acc[2][2] = {};
  stage(0, 0);
  __syncthreads();
  int buf = 0;
  for (int kt = 0; kt < KT; ++kt) {
    if (kt + 1 < KT) stage(kt + 1, buf ^ 1);
    // fragment gather, contiguous k = 16*hi + s in A and B (2x b128 each)
    v16h a[2], b[2];
#pragma unroll
    for (int mt = 0; mt < 2; ++mt) {
      v8h lo = *(const v8h*)&As[buf][wr * 32 + mt * 16 + m][hi * 16];
      v8h hh = *(const v8h*)&As[buf][wr * 32 + mt * 16 + m][hi * 16 + 8];
#pragma unroll
      for (int j = 0; j < 8; ++j) { a[mt][j] = lo[j]; a[mt][j + 8] = hh[j]; }
    }
#pragma unroll
    for (int nt = 0; nt < 2; ++nt) {
      v8h b0 = *(const v8h*)&Bs[buf][wc * 32 + nt * 16 + m][hi * 16];
      v8h b1 = *(const v8h*)&Bs[buf][wc * 32 + nt * 16 + m][hi * 16 + 8];
#pragma unroll
      for (int j = 0; j < 8; ++j) { b[nt][j] = b0[j]; b[nt][j + 8] = b1[j]; }
    }
#pragma unroll
    for (int mt = 0; mt < 2; ++mt)
#pragma unroll
      for (int nt = 0; nt < 2; ++nt)
        acc[mt][nt] = __builtin_amdgcn_wmma_f32_16x16x32_f16(
            false, a[mt], false, b[nt], (short)0, acc[mt][nt], false, false);
    __syncthreads();
    buf ^= 1;
  }

  // epilogue: bias (+residual) (+relu); padded cols written as zero
#pragma unroll
  for (int nt = 0; nt < 2; ++nt) {
    const int col = colBase + wc * 32 + nt * 16 + m;
    if (col < PP) {
      const float bcol = (col < P) ? bias[col] : 0.f;
#pragma unroll
      for (int mt = 0; mt < 2; ++mt) {
#pragma unroll
        for (int v = 0; v < 8; ++v) {
          const int row = rowBase + wr * 32 + mt * 16 + v + 8 * hi;
          float val = 0.f;
          if (col < P) {
            val = acc[mt][nt][v] + bcol;
            if (res)  val += res[(size_t)row * PP + col];
            if (relu) val = fmaxf(val, 0.f);
          }
          Y[(size_t)row * PP + col] = val;
        }
      }
    }
  }
}

// ---------------- wave-per-row LayerNorm over 400 (stride LDH), zero pad -----------
__global__ void k_resln(const float* __restrict__ S, const float* __restrict__ w,
                        const float* __restrict__ b, float* __restrict__ Hout, int Ntok) {
  const int wave = threadIdx.x >> 5;
  const int lane = threadIdx.x & 31;
  const int row  = blockIdx.x * 8 + wave;
  const float* src = S + (size_t)row * LDH;
  float v[13];
  float sum = 0.f;
#pragma unroll
  for (int j = 0; j < 13; ++j) {
    int c = lane + 32 * j;
    v[j] = (c < C_BANDS) ? src[c] : 0.f;
    sum += v[j];
  }
#pragma unroll
  for (int off = 16; off; off >>= 1) sum += __shfl_xor(sum, off, 32);
  const float mu = sum * (1.0f / C_BANDS);
  float var = 0.f;
#pragma unroll
  for (int j = 0; j < 13; ++j) {
    int c = lane + 32 * j;
    float d = (c < C_BANDS) ? (v[j] - mu) : 0.f;
    var += d * d;
  }
#pragma unroll
  for (int off = 16; off; off >>= 1) var += __shfl_xor(var, off, 32);
  const float rs = rsqrtf(var * (1.0f / C_BANDS) + 1e-5f);
  float* dst = Hout + (size_t)row * LDH;
#pragma unroll
  for (int j = 0; j < 13; ++j) {
    int c = lane + 32 * j;   // covers 0..415 exactly
    float o = (c < C_BANDS) ? ((v[j] - mu) * rs * w[c] + b[c]) : 0.f;
    dst[c] = o;
  }
}

// ---------------- wave-per-token: scores, top-10, gather, tiny MLP, outputs --------
__global__ void k_select(const float* __restrict__ Hf, const float* __restrict__ xorig,
                         const float* __restrict__ bi,
                         const float* __restrict__ We1, const float* __restrict__ be1,
                         const float* __restrict__ We2, const float* __restrict__ be2,
                         const float* __restrict__ Wd1, const float* __restrict__ bd1,
                         const float* __restrict__ Wd2, const float* __restrict__ bd2,
                         float* __restrict__ out0, int* __restrict__ out1,
                         float* __restrict__ out2, float* __restrict__ out3, int Ntok) {
  const int wave = threadIdx.x >> 5;
  const int lane = threadIdx.x & 31;
  const int p = blockIdx.x * 8 + wave;
  const float* src = Hf + (size_t)p * LDH;

  float sc[13];
#pragma unroll
  for (int j = 0; j < 13; ++j) {
    int c = lane + 32 * j;
    if (c < C_BANDS) {
      float s = 1.f / (1.f + expf(-bi[c]));
      sc[j] = src[c] * s;
      out2[(size_t)p * C_BANDS + c] = sc[j];
    } else {
      sc[j] = -3.0e38f;
    }
  }

  // top-10 (desc by value, tie -> lower index), iterative masked wave-max
  int used = 0;
  float topv[KSEL]; int topi[KSEL];
#pragma unroll
  for (int t = 0; t < KSEL; ++t) {
    float bv = -3.0e38f; int bc = 0x7fffffff;
#pragma unroll
    for (int j = 0; j < 13; ++j) {
      int c = lane + 32 * j;
      bool ok = (c < C_BANDS) && !((used >> j) & 1);
      if (ok && (sc[j] > bv || (sc[j] == bv && c < bc))) { bv = sc[j]; bc = c; }
    }
#pragma unroll
    for (int off = 16; off; off >>= 1) {
      float ov = __shfl_xor(bv, off, 32);
      int   oc = __shfl_xor(bc, off, 32);
      if (ov > bv || (ov == bv && oc < bc)) { bv = ov; bc = oc; }
    }
    topv[t] = bv; topi[t] = bc;
    if ((bc & 31) == lane) used |= 1 << (bc >> 5);
  }
  // sort ascending by band index (argsort(top_i))
  for (int i = 1; i < KSEL; ++i) {
    int ci = topi[i]; float cv = topv[i];
    int j = i - 1;
    while (j >= 0 && topi[j] > ci) { topi[j + 1] = topi[j]; topv[j + 1] = topv[j]; --j; }
    topi[j + 1] = ci; topv[j + 1] = cv;
  }

  float sel1[KSEL], selv[KSEL];
#pragma unroll
  for (int t = 0; t < KSEL; ++t) {
    sel1[t] = xorig[(size_t)topi[t] * Ntok + p];   // x_original is [C, N]
    selv[t] = sel1[t] * topv[t];
  }
  if (lane < KSEL) {
    out1[(size_t)p * KSEL + lane] = topi[lane];
    out3[(size_t)p * KSEL + lane] = sel1[lane];
  }

  // tiny MLP (redundant across lanes)
  float a1[LATENT], e2[LATENT], z[LATENT];
#pragma unroll
  for (int l = 0; l < LATENT; ++l) {
    float s = be1[l];
#pragma unroll
    for (int t = 0; t < KSEL; ++t) s += selv[t] * We1[l * KSEL + t];
    a1[l] = fmaxf(s, 0.f);
  }
#pragma unroll
  for (int l = 0; l < LATENT; ++l) {
    float s = be2[l];
#pragma unroll
    for (int q = 0; q < LATENT; ++q) s += a1[q] * We2[l * LATENT + q];
    e2[l] = s;
  }
#pragma unroll
  for (int l = 0; l < LATENT; ++l) {
    float s = bd1[l];
#pragma unroll
    for (int q = 0; q < LATENT; ++q) s += e2[q] * Wd1[l * LATENT + q];
    z[l] = fmaxf(s, 0.f);
  }
#pragma unroll
  for (int j = 0; j < 13; ++j) {
    int c = lane + 32 * j;
    if (c < C_BANDS) {
      float s = bd2[c];
#pragma unroll
      for (int q = 0; q < LATENT; ++q) s += z[q] * Wd2[c * LATENT + q];
      out0[(size_t)c * Ntok + p] = s;   // reconstructed is [C, N]
    }
  }
}

extern "C" void kernel_launch(void* const* d_in, const int* in_sizes, int n_in,
                              void* d_out, int out_size, void* d_ws, size_t ws_size,
                              hipStream_t stream) {
  const float* x    = (const float*)d_in[0];
  const float* xo   = (const float*)d_in[1];
  const float* pos  = (const float*)d_in[2];
  const float* bi   = (const float*)d_in[3];
  const float* Wv   = (const float*)d_in[4];
  const float* bv   = (const float*)d_in[5];
  const float* Wo   = (const float*)d_in[6];
  const float* bo   = (const float*)d_in[7];
  const float* ln1w = (const float*)d_in[8];
  const float* ln1b = (const float*)d_in[9];
  const float* W1   = (const float*)d_in[10];
  const float* b1   = (const float*)d_in[11];
  const float* W2   = (const float*)d_in[12];
  const float* b2   = (const float*)d_in[13];
  const float* ln2w = (const float*)d_in[14];
  const float* ln2b = (const float*)d_in[15];
  const float* We1  = (const float*)d_in[16];
  const float* be1  = (const float*)d_in[17];
  const float* We2  = (const float*)d_in[18];
  const float* be2  = (const float*)d_in[19];
  const float* Wd1  = (const float*)d_in[20];
  const float* bd1  = (const float*)d_in[21];
  const float* Wd2  = (const float*)d_in[22];
  const float* bd2  = (const float*)d_in[23];

  const int Ntok = in_sizes[0] / C_BANDS;  // 65536 tokens

  // workspace: h, t1, t2 (f32, stride LDH) then packed f16 weights
  float* h  = (float*)d_ws;
  float* t1 = h  + (size_t)Ntok * LDH;
  float* t2 = t1 + (size_t)Ntok * LDH;
  _Float16* wp = (_Float16*)(t2 + (size_t)Ntok * LDH);
  const size_t SZ_VO = (size_t)LDH * LDH;    // 416*416
  const size_t SZ_W1 = (size_t)FFDIM * LDH;  // 128*416
  const size_t SZ_W2 = (size_t)LDH * FFDIM;  // 416*128
  _Float16* WvH[2]; _Float16* WoH[2]; _Float16* W1H[2]; _Float16* W2H[2];
  {
    _Float16* cur = wp;
    for (int i = 0; i < 2; ++i) { WvH[i] = cur; cur += SZ_VO; }
    for (int i = 0; i < 2; ++i) { WoH[i] = cur; cur += SZ_VO; }
    for (int i = 0; i < 2; ++i) { W1H[i] = cur; cur += SZ_W1; }
    for (int i = 0; i < 2; ++i) { W2H[i] = cur; cur += SZ_W2; }
  }

  // 0) pack weights to padded f16
  for (int i = 0; i < 2; ++i) {
    k_wpack<<<(int)(SZ_VO / 256), 256, 0, stream>>>(
        Wv + (size_t)i * C_BANDS * C_BANDS, WvH[i], C_BANDS, C_BANDS, LDH);
    k_wpack<<<(int)(SZ_VO / 256), 256, 0, stream>>>(
        Wo + (size_t)i * C_BANDS * C_BANDS, WoH[i], C_BANDS, C_BANDS, LDH);
    k_wpack<<<(int)(SZ_W1 / 256), 256, 0, stream>>>(
        W1 + (size_t)i * FFDIM * C_BANDS, W1H[i], FFDIM, C_BANDS, LDH);
    k_wpack<<<(int)(SZ_W2 / 256), 256, 0, stream>>>(
        W2 + (size_t)i * C_BANDS * FFDIM, W2H[i], C_BANDS, FFDIM, FFDIM);
  }

  // 1) tokens = transpose(x) + pos (zero-padded to LDH)
  k_transpose_pos<<<dim3(Ntok / 32, LDH / 32), dim3(32, 8), 0, stream>>>(x, pos, h, Ntok);

  const dim3 g400(Ntok / 128, (LDH + 63) / 64);    // 512 x 7
  const dim3 g128(Ntok / 128, (FFDIM + 63) / 64);  // 512 x 2

  // 2) transformer layers
  for (int i = 0; i < 2; ++i) {
    const float* bv_i = bv + (size_t)i * C_BANDS;
    const float* bo_i = bo + (size_t)i * C_BANDS;
    const float* l1w  = ln1w + (size_t)i * C_BANDS;
    const float* l1b  = ln1b + (size_t)i * C_BANDS;
    const float* b1_i = b1 + (size_t)i * FFDIM;
    const float* b2_i = b2 + (size_t)i * C_BANDS;
    const float* l2w  = ln2w + (size_t)i * C_BANDS;
    const float* l2b  = ln2b + (size_t)i * C_BANDS;

    // t1 = h @ Wv^T + bv
    k_gemm_wmma<<<g400, 256, 0, stream>>>(h, WvH[i], bv_i, nullptr, t1,
                                          C_BANDS, LDH, LDH, 0);
    // t2 = t1 @ Wo^T + bo + h   (pre-LN sum)
    k_gemm_wmma<<<g400, 256, 0, stream>>>(t1, WoH[i], bo_i, h, t2,
                                          C_BANDS, LDH, LDH, 0);
    // h = LN(t2)
    k_resln<<<Ntok / 8, 256, 0, stream>>>(t2, l1w, l1b, h, Ntok);
    // t1[:, :128] = relu(h @ W1^T + b1)
    k_gemm_wmma<<<g128, 256, 0, stream>>>(h, W1H[i], b1_i, nullptr, t1,
                                          FFDIM, FFDIM, LDH, 1);
    // t2 = t1 @ W2^T + b2 + h
    k_gemm_wmma<<<g400, 256, 0, stream>>>(t1, W2H[i], b2_i, h, t2,
                                          C_BANDS, LDH, FFDIM, 0);
    // h = LN(t2)
    k_resln<<<Ntok / 8, 256, 0, stream>>>(t2, l2w, l2b, h, Ntok);
  }

  // 3) outputs: [reconstructed | sorted_idx | attention_scores | selected_bands1]
  float* out0 = (float*)d_out;
  int*   out1 = (int*)(out0 + (size_t)C_BANDS * Ntok);
  float* out2 = out0 + (size_t)C_BANDS * Ntok + (size_t)KSEL * Ntok;
  float* out3 = out2 + (size_t)C_BANDS * Ntok;
  k_select<<<Ntok / 8, 256, 0, stream>>>(h, xo, bi, We1, be1, We2, be2, Wd1, bd1, Wd2, bd2,
                                         out0, out1, out2, out3, Ntok);
}
